// CRF_89421219103173
// MI455X (gfx1250) — matrix operand order repaired and checked by reading, not compiled, not run on hardware
//
#include <hip/hip_runtime.h>
#include <hip/hip_bf16.h>

typedef __attribute__((ext_vector_type(16))) _Float16 v16h;
typedef __attribute__((ext_vector_type(8)))  _Float16 h8;
typedef __attribute__((ext_vector_type(8)))  float    v8f;
typedef __attribute__((ext_vector_type(4)))  float    f4;

#define NT 128   // tags
#define SS 1024  // sequence length
#define NB 256   // batch
#define ROWS 16  // batch rows per workgroup

#define LOG2E 1.4426950408889634f
#define LN2   0.6931471805599453f

// Raw hardware transcendentals: args are range-safe by construction
// (exp input <= 0 after max-subtraction; log input in [~6e-3, ~7e3]).
__device__ __forceinline__ float fast_exp(float x) {
  return __builtin_amdgcn_exp2f(x * LOG2E);
}
__device__ __forceinline__ float fast_log(float x) {
  return __builtin_amdgcn_logf(x) * LN2;
}

// -------------------------------------------------------------------------
// Kernel 1: CRF forward scan. One WG = 16 batch rows, 8 waves (256 threads).
// Each wave owns a 16-wide column tile of alpha; E = exp(trans) lives in
// per-wave B-fragment registers for the whole scan. Per step:
//   P = exp(alpha - rowmax)  (written to LDS directly in WMMA A-layout)
//   C = P @ E  via 4 chained v_wmma_f32_16x16x32_f16  (K = 128)
//   alpha' = mask ? rowmax + log(C) + emit : alpha
// -------------------------------------------------------------------------
__global__ __launch_bounds__(256) void crf_forward_kernel(
    const float* __restrict__ emissions,
    const float* __restrict__ masks,
    const float* __restrict__ trans,
    const float* __restrict__ start_t,
    const float* __restrict__ end_t,
    float* __restrict__ logz)
{
  __shared__ __align__(16) float    s_alpha[ROWS][NT];  // 8 KB
  __shared__ __align__(32) _Float16 s_A[4][32][16];     // 4 KB, A-fragment layout
  __shared__ __align__(16) float    s_rowmax[ROWS];
  __shared__ __align__(16) float    s_mask[ROWS];

  const int tid  = threadIdx.x;
  const int lane = tid & 31;
  const int hi   = lane >> 4;       // 0/1
  const int ln   = lane & 15;
  const int b0   = blockIdx.x * ROWS;
  const int col  = (tid >> 5) * 16 + ln;  // tag column owned by this lane (B/C layout)

  // P-phase role: thread t handles row m_row, 8 contiguous K values at cbase.
  const int m_row = tid >> 4;            // 0..15
  const int cbase = (tid & 15) * 8;      // K base, multiple of 8
  const int chunk = cbase >> 5;          // which 32-wide K chunk
  const int qq    = (cbase & 31) >> 3;   // 8-group within chunk
  const int alane = (qq >> 1) * 16 + m_row;
  const int ebase = (qq & 1) * 8;        // element offset inside v16h

  // init alpha = start_transitions broadcast over rows
#pragma unroll
  for (int e = 0; e < 8; ++e) s_alpha[m_row][cbase + e] = start_t[cbase + e];

  // B fragments: E[i,j] = exp(trans[i,j]) for this wave's 16 columns, K=0..127.
  // B layout (32x16 f16): lane holds col = ln, element e -> K = hi*16 + e.
  v16h bfrag[4];
#pragma unroll
  for (int c = 0; c < 4; ++c) {
#pragma unroll
    for (int e = 0; e < 16; ++e) {
      const int K = c * 32 + hi * 16 + e;
      bfrag[c][e] = (_Float16)fast_exp(trans[K * NT + col]);
    }
  }

  // alpha in C-layout registers: row r + 8*hi, col = col
  float areg[8];
#pragma unroll
  for (int r = 0; r < 8; ++r) areg[r] = start_t[col];

  __syncthreads();

  for (int s = 0; s < SS; ++s) {
    // Issue emission loads early so global latency overlaps the P-phase.
    float em[8];
#pragma unroll
    for (int r = 0; r < 8; ++r) {
      const int b = b0 + r + 8 * hi;
      em[r] = emissions[(b * SS + s) * NT + col];
    }
    // One thread per row stages the step mask into LDS (broadcast later).
    if ((tid & 15) == 1) s_mask[m_row] = masks[(b0 + m_row) * SS + s];

    // ---- P phase: rowmax + exp, write A fragments ----
    f4 a0 = *(const f4*)&s_alpha[m_row][cbase];
    f4 a1 = *(const f4*)&s_alpha[m_row][cbase + 4];
    float vmax = fmaxf(fmaxf(fmaxf(a0.x, a0.y), fmaxf(a0.z, a0.w)),
                       fmaxf(fmaxf(a1.x, a1.y), fmaxf(a1.z, a1.w)));
#pragma unroll
    for (int off = 8; off >= 1; off >>= 1)
      vmax = fmaxf(vmax, __shfl_xor(vmax, off, 16));

    h8 ph;
    ph[0] = (_Float16)fast_exp(a0.x - vmax);
    ph[1] = (_Float16)fast_exp(a0.y - vmax);
    ph[2] = (_Float16)fast_exp(a0.z - vmax);
    ph[3] = (_Float16)fast_exp(a0.w - vmax);
    ph[4] = (_Float16)fast_exp(a1.x - vmax);
    ph[5] = (_Float16)fast_exp(a1.y - vmax);
    ph[6] = (_Float16)fast_exp(a1.z - vmax);
    ph[7] = (_Float16)fast_exp(a1.w - vmax);
    *(h8*)&s_A[chunk][alane][ebase] = ph;
    if ((tid & 15) == 0) s_rowmax[m_row] = vmax;
    __syncthreads();

    // ---- WMMA phase: C = P @ E over K=128 (4 chunks of 32) ----
    v16h fA0 = *(const v16h*)&s_A[0][lane][0];
    v16h fA1 = *(const v16h*)&s_A[1][lane][0];
    v16h fA2 = *(const v16h*)&s_A[2][lane][0];
    v16h fA3 = *(const v16h*)&s_A[3][lane][0];
    v8f acc = {};
    acc = __builtin_amdgcn_wmma_f32_16x16x32_f16(false, fA0, false, bfrag[0],
                                                 (short)0, acc, false, false);
    acc = __builtin_amdgcn_wmma_f32_16x16x32_f16(false, fA1, false, bfrag[1],
                                                 (short)0, acc, false, false);
    acc = __builtin_amdgcn_wmma_f32_16x16x32_f16(false, fA2, false, bfrag[2],
                                                 (short)0, acc, false, false);
    acc = __builtin_amdgcn_wmma_f32_16x16x32_f16(false, fA3, false, bfrag[3],
                                                 (short)0, acc, false, false);

    // ---- alpha update (branchless; rowmax/mask via contiguous f4 LDS loads) ----
    f4 rm0 = *(const f4*)&s_rowmax[8 * hi];
    f4 rm1 = *(const f4*)&s_rowmax[8 * hi + 4];
    f4 mk0 = *(const f4*)&s_mask[8 * hi];
    f4 mk1 = *(const f4*)&s_mask[8 * hi + 4];
    float rm[8] = {rm0.x, rm0.y, rm0.z, rm0.w, rm1.x, rm1.y, rm1.z, rm1.w};
    float mk[8] = {mk0.x, mk0.y, mk0.z, mk0.w, mk1.x, mk1.y, mk1.z, mk1.w};
#pragma unroll
    for (int r = 0; r < 8; ++r) {
      const float nxt = rm[r] + fast_log(acc[r]) + em[r];
      areg[r] = (mk[r] > 0.0f) ? nxt : areg[r];
      s_alpha[r + 8 * hi][col] = areg[r];
    }
    __syncthreads();
  }

  // ---- log_z = logsumexp(alpha + end_transitions) per row ----
  f4 a0 = *(const f4*)&s_alpha[m_row][cbase];
  f4 a1 = *(const f4*)&s_alpha[m_row][cbase + 4];
  float w0 = a0.x + end_t[cbase + 0];
  float w1 = a0.y + end_t[cbase + 1];
  float w2 = a0.z + end_t[cbase + 2];
  float w3 = a0.w + end_t[cbase + 3];
  float w4 = a1.x + end_t[cbase + 4];
  float w5 = a1.y + end_t[cbase + 5];
  float w6 = a1.z + end_t[cbase + 6];
  float w7 = a1.w + end_t[cbase + 7];
  float vmax = fmaxf(fmaxf(fmaxf(w0, w1), fmaxf(w2, w3)),
                     fmaxf(fmaxf(w4, w5), fmaxf(w6, w7)));
#pragma unroll
  for (int off = 8; off >= 1; off >>= 1)
    vmax = fmaxf(vmax, __shfl_xor(vmax, off, 16));
  float ssum = fast_exp(w0 - vmax) + fast_exp(w1 - vmax) + fast_exp(w2 - vmax) +
               fast_exp(w3 - vmax) + fast_exp(w4 - vmax) + fast_exp(w5 - vmax) +
               fast_exp(w6 - vmax) + fast_exp(w7 - vmax);
#pragma unroll
  for (int off = 8; off >= 1; off >>= 1)
    ssum += __shfl_xor(ssum, off, 16);
  if ((tid & 15) == 0) logz[b0 + m_row] = vmax + fast_log(ssum);
}

// -------------------------------------------------------------------------
// Kernel 2: gold-path score per batch row. One block per row.
// -------------------------------------------------------------------------
__global__ __launch_bounds__(256) void crf_score_kernel(
    const float* __restrict__ emissions, const float* __restrict__ masks,
    const int* __restrict__ tags, const float* __restrict__ trans,
    const float* __restrict__ start_t, const float* __restrict__ end_t,
    float* __restrict__ score)
{
  const int b = blockIdx.x;
  const int t = threadIdx.x;
  float acc = 0.0f, msum = 0.0f;
  for (int s = t; s < SS; s += 256) {
    const int   tg = tags[b * SS + s];
    const float mv = masks[b * SS + s];
    msum += mv;
    if (s < SS - 1) acc += emissions[(b * SS + s) * NT + tg] * mv;
    if (s >= 1)     acc += trans[tags[b * SS + s - 1] * NT + tg] * mv;
  }
#pragma unroll
  for (int off = 16; off >= 1; off >>= 1) {
    acc  += __shfl_down(acc,  off, 32);
    msum += __shfl_down(msum, off, 32);
  }
  __shared__ float r_acc[8], r_ms[8];
  if ((t & 31) == 0) { r_acc[t >> 5] = acc; r_ms[t >> 5] = msum; }
  __syncthreads();
  if (t == 0) {
    float ta = 0.0f, tm = 0.0f;
    for (int i = 0; i < 8; ++i) { ta += r_acc[i]; tm += r_ms[i]; }
    const int   tg0 = tags[b * SS];
    const int   tgl = tags[b * SS + SS - 1];
    const float ml  = masks[b * SS + SS - 1];
    const int last_ix = (int)fmaxf(tm - 1.0f, 0.0f);
    float sc = start_t[tg0] + ta;
    sc += emissions[(b * SS + last_ix) * NT + tgl] * ml;
    sc += end_t[tgl] * ml;
    score[b] = sc;
  }
}

// -------------------------------------------------------------------------
// Kernel 3: deterministic final reduction: mean(log_z - score).
// -------------------------------------------------------------------------
__global__ __launch_bounds__(256) void crf_final_kernel(
    const float* __restrict__ logz, const float* __restrict__ score,
    float* __restrict__ out)
{
  const int t = threadIdx.x;
  float v = logz[t] - score[t];
#pragma unroll
  for (int off = 16; off >= 1; off >>= 1) v += __shfl_down(v, off, 32);
  __shared__ float r[8];
  if ((t & 31) == 0) r[t >> 5] = v;
  __syncthreads();
  if (t == 0) {
    float sum = 0.0f;
    for (int i = 0; i < 8; ++i) sum += r[i];
    out[0] = sum * (1.0f / (float)NB);
  }
}

extern "C" void kernel_launch(void* const* d_in, const int* in_sizes, int n_in,
                              void* d_out, int out_size, void* d_ws, size_t ws_size,
                              hipStream_t stream) {
  const float* emissions = (const float*)d_in[0];   // (256,1024,128) f32
  const float* masks     = (const float*)d_in[1];   // (256,1024)     f32
  const int*   tags      = (const int*)  d_in[2];   // (256,1024)     int
  const float* trans     = (const float*)d_in[3];   // (128,128)      f32
  const float* start_t   = (const float*)d_in[4];   // (128,)         f32
  const float* end_t     = (const float*)d_in[5];   // (128,)         f32
  float* out  = (float*)d_out;
  float* logz  = (float*)d_ws;        // 256 floats
  float* score = logz + NB;           // 256 floats

  crf_forward_kernel<<<NB / ROWS, 256, 0, stream>>>(emissions, masks, trans,
                                                    start_t, end_t, logz);
  crf_score_kernel<<<NB, 256, 0, stream>>>(emissions, masks, tags, trans,
                                           start_t, end_t, score);
  crf_final_kernel<<<1, 256, 0, stream>>>(logz, score, out);
}
